// MPNNModel_11519102288405
// MI455X (gfx1250) — matrix operand order repaired and checked by reading, not compiled, not run on hardware
//
#include <hip/hip_runtime.h>
#include <stdint.h>

typedef __attribute__((ext_vector_type(16))) _Float16 v16h;
typedef __attribute__((ext_vector_type(2)))  _Float16 v2h;
typedef __attribute__((ext_vector_type(8)))  float    v8f;

#define NN     100000
#define NE     1600000
#define EMB    64
#define LAYERS 4

// Single-instruction ReLU: v_med3_f32(x, 0, FLT_MAX). Finite upper bound keeps
// InstCombine from folding back to canonicalize+fmaxnum (which it does for +inf).
__device__ __forceinline__ float relu(float x) {
  return __builtin_amdgcn_fmed3f(x, 0.0f, 3.402823466e+38f);
}

// ---------------------------------------------------------------------------
// WMMA helper: D = A(16x32 f16) * B(32x16 f16) + C(16x16 f32)
// ---------------------------------------------------------------------------
__device__ __forceinline__ v8f wmma_f16(v16h a, v16h b, v8f c) {
  return __builtin_amdgcn_wmma_f32_16x16x32_f16(
      /*neg_a=*/false, a, /*neg_b=*/false, b,
      /*c_mod=*/(short)0, c, /*reuse_a=*/false, /*reuse_b=*/false);
}

// A-fragment (16x32 f16) from an f16 row (global h16 or LDS hidden).
// Lane l (pair l, l+16 share row M=l&15): halves 0..7  <- row[colbase..+7]
//                                         halves 8..15 <- row[colbase+16..+23]
__device__ __forceinline__ v16h load_a_f16row(const _Float16* row, int colbase) {
  union { int4 q[2]; v16h v; } u;
  u.q[0] = *reinterpret_cast<const int4*>(row + colbase);        // 8 halves = 16B
  u.q[1] = *reinterpret_cast<const int4*>(row + colbase + 16);   // 8 halves = 16B
  return u.v;
}

// A-fragment from an f32 row (aggr in upd kernel) with on-the-fly cvt.
__device__ __forceinline__ v16h load_a_f32row(const float* __restrict__ row, int colbase) {
  float4 u0 = *reinterpret_cast<const float4*>(row + colbase);
  float4 u1 = *reinterpret_cast<const float4*>(row + colbase + 4);
  float4 u2 = *reinterpret_cast<const float4*>(row + colbase + 16);
  float4 u3 = *reinterpret_cast<const float4*>(row + colbase + 20);
  v16h a;
  a[0]=(_Float16)u0.x; a[1]=(_Float16)u0.y; a[2]=(_Float16)u0.z; a[3]=(_Float16)u0.w;
  a[4]=(_Float16)u1.x; a[5]=(_Float16)u1.y; a[6]=(_Float16)u1.z; a[7]=(_Float16)u1.w;
  a[8]=(_Float16)u2.x; a[9]=(_Float16)u2.y; a[10]=(_Float16)u2.z; a[11]=(_Float16)u2.w;
  a[12]=(_Float16)u3.x; a[13]=(_Float16)u3.y; a[14]=(_Float16)u3.z; a[15]=(_Float16)u3.w;
  return a;
}

// B-fragment: 32 contiguous bytes per lane, pre-swizzled in LDS.
__device__ __forceinline__ v16h load_b_frag(const _Float16* p) {
  union { int4 q[2]; v16h v; } u;
  u.q[0] = *reinterpret_cast<const int4*>(p);
  u.q[1] = *reinterpret_cast<const int4*>(p + 8);
  return u.v;
}

// Stage an f32 weight matrix [Krows x 64] into LDS as f16 B-fragments.
// Fragment (kt,nt): lane l holds col N = nt*16+(l&15), K = kt*32+(h>>3)*16+(l>>4)*8+(h&7).
// Optional bias row: staged at K == Krows (activated by A-element 1.0).
__device__ __forceinline__ void stage_w(const float* __restrict__ W,
                                        const float* __restrict__ bias, // may be null
                                        int Krows, int Kfrags, _Float16* sOut,
                                        int tid, int nthreads) {
  int total = Kfrags * 4 * 512;
  for (int e = tid; e < total; e += nthreads) {
    int frag = e >> 9;
    int idx  = e & 511;
    int lane = idx >> 4;
    int hh   = idx & 15;
    int kt = frag >> 2, nt = frag & 3;
    int K   = kt * 32 + ((hh >> 3) << 4) + ((lane >> 4) << 3) + (hh & 7);
    int col = nt * 16 + (lane & 15);
    float v = 0.0f;
    if (K < Krows)               v = W[(size_t)K * 64 + col];
    else if (bias && K == Krows) v = bias[col];
    sOut[(size_t)frag * 512 + lane * 16 + hh] = (_Float16)v;
  }
}

// cvt pair to f16 then packed ReLU (v_pk_max_num_f16; cvt outputs are canonical
// so no canonicalize is needed). Equivalent to relu-then-cvt (monotone rounding).
__device__ __forceinline__ v2h relu_pair_f16(float a, float b) {
  v2h p;
  p[0] = (_Float16)a;
  p[1] = (_Float16)b;
  v2h z = {(_Float16)0.0f, (_Float16)0.0f};
  return __builtin_elementwise_max(p, z);
}

// ---------------------------------------------------------------------------
// Edge/message kernel: gather (f16) -> MLP(132->64->64) -> atomic scatter-add
// One wave = 16 edges. Persistent grid-stride over E/16 tiles.
// ---------------------------------------------------------------------------
__global__ __launch_bounds__(256) void msg_kernel(
    const _Float16* __restrict__ h16, const long long* __restrict__ ei,
    const float* __restrict__ ea,
    const float* __restrict__ w1, const float* __restrict__ b1,
    const float* __restrict__ w2, const float* __restrict__ b2,
    float* __restrict__ aggr) {
  __shared__ __align__(16) _Float16 sW1[5 * 4 * 512];   // K=132 (+bias row 132) -> 5 K-tiles
  __shared__ __align__(16) _Float16 sW2[2 * 4 * 512];   // K=64 -> 2 K-tiles
  __shared__ float sB2[EMB];
  __shared__ __align__(16) _Float16 sHid[8][16 * EMB];  // per-wave hidden tile
  __shared__ unsigned sDst[8][16];

  const int tid = threadIdx.x;
  stage_w(w1, b1, 132, 5, sW1, tid, 256);   // bias folded at K=132
  stage_w(w2, nullptr, 64, 2, sW2, tid, 256);
  if (tid < EMB) sB2[tid] = b2[tid];
  __syncthreads();

  const int wv = tid >> 5, lane = tid & 31;
  const int lo = lane & 15, hi = lane >> 4;
  const int base = hi * 8;
  const int numTiles = NE / 16;
  const int stride = gridDim.x * 8;

  // Edge-attr A fragment: halves 5..15 are invariantly zero (hi==0 rewrites 0..4
  // each tile; hi==1 lanes contribute a zero fragment). Hoisted out of the loop.
  v16h aE = {};

  for (int tile = blockIdx.x * 8 + wv; tile < numTiles; tile += stride) {
    const int erow = tile * 16 + lo;
    const long long jsrc = ei[erow];
    const long long idst = ei[NE + erow];
    if (lane < 16) sDst[wv][lo] = (unsigned)idst;
    const _Float16* hrow_i = h16 + (size_t)idst * EMB;
    const _Float16* hrow_j = h16 + (size_t)jsrc * EMB;

    // prefetch next tile's edge data (global_prefetch_b8)
    {
      int ntile = tile + stride;
      if (ntile < numTiles) {
        int nerow = ntile * 16 + lo;
        __builtin_prefetch(&ei[nerow], 0, 3);
        __builtin_prefetch(&ei[NE + nerow], 0, 3);
        __builtin_prefetch(&ea[(size_t)nerow * 4], 0, 3);
      }
    }

    // ---- GEMM1: [16 x 133] x [133 x 64]; bias via K=132, C starts at inline 0
    v8f acc[4] = {};
#pragma unroll
    for (int kt = 0; kt < 2; ++kt) {                 // h_i part (K 0..63)
      v16h a = load_a_f16row(hrow_i, kt * 32 + base);
#pragma unroll
      for (int nt = 0; nt < 4; ++nt)
        acc[nt] = wmma_f16(a, load_b_frag(&sW1[(size_t)(kt * 4 + nt) * 512 + lane * 16]), acc[nt]);
    }
#pragma unroll
    for (int kt = 0; kt < 2; ++kt) {                 // h_j part (K 64..127)
      v16h a = load_a_f16row(hrow_j, kt * 32 + base);
#pragma unroll
      for (int nt = 0; nt < 4; ++nt)
        acc[nt] = wmma_f16(a, load_b_frag(&sW1[(size_t)((kt + 2) * 4 + nt) * 512 + lane * 16]), acc[nt]);
    }
    {                                                // edge_attr (K 128..131) + bias one-hot (K 132)
      if (hi == 0) {
        float4 v = *reinterpret_cast<const float4*>(ea + (size_t)erow * 4);
        aE[0] = (_Float16)v.x; aE[1] = (_Float16)v.y;
        aE[2] = (_Float16)v.z; aE[3] = (_Float16)v.w;
        aE[4] = (_Float16)1.0f;                      // activates bias row K=132
      }
#pragma unroll
      for (int nt = 0; nt < 4; ++nt)
        acc[nt] = wmma_f16(aE, load_b_frag(&sW1[(size_t)(16 + nt) * 512 + lane * 16]), acc[nt]);
    }

    // ---- cvt + packed-f16 ReLU -> hidden tile in LDS (C-layout -> A-layout)
#pragma unroll
    for (int nt = 0; nt < 4; ++nt)
#pragma unroll
      for (int rp = 0; rp < 4; ++rp) {
        v2h p = relu_pair_f16(acc[nt][2 * rp], acc[nt][2 * rp + 1]);
        sHid[wv][(2 * rp     + 8 * hi) * EMB + nt * 16 + lo] = p[0];
        sHid[wv][(2 * rp + 1 + 8 * hi) * EMB + nt * 16 + lo] = p[1];
      }

    // ---- GEMM2: [16 x 64] x [64 x 64], bias-initialized accumulators
    v8f acc2[4];
#pragma unroll
    for (int nt = 0; nt < 4; ++nt) {
      float bv = sB2[nt * 16 + lo];
#pragma unroll
      for (int r = 0; r < 8; ++r) acc2[nt][r] = bv;
    }
#pragma unroll
    for (int kt = 0; kt < 2; ++kt) {
      v16h a = load_a_f16row(&sHid[wv][lo * EMB], kt * 32 + base);
#pragma unroll
      for (int nt = 0; nt < 4; ++nt)
        acc2[nt] = wmma_f16(a, load_b_frag(&sW2[(size_t)(kt * 4 + nt) * 512 + lane * 16]), acc2[nt]);
    }

    // ---- ReLU (v_med3) + scatter-add into aggr[dst]
#pragma unroll
    for (int r = 0; r < 8; ++r) {
      unsigned node = sDst[wv][r + 8 * hi];
      float* arow = aggr + (size_t)node * EMB;
#pragma unroll
      for (int nt = 0; nt < 4; ++nt)
        unsafeAtomicAdd(arow + nt * 16 + lo, relu(acc2[nt][r]));  // global_atomic_add_f32
    }
  }
}

// ---------------------------------------------------------------------------
// Node update kernel: [h | aggr] -> MLP(128->64->64), residual h += upd,
// refresh f16 shadow h16. One wave = 16 nodes.
// ---------------------------------------------------------------------------
__global__ __launch_bounds__(256) void upd_kernel(
    float* __restrict__ h, _Float16* __restrict__ h16,
    const float* __restrict__ aggr,
    const float* __restrict__ w1, const float* __restrict__ b1,
    const float* __restrict__ w2, const float* __restrict__ b2) {
  __shared__ __align__(16) _Float16 sW1[4 * 4 * 512];   // K=128 -> 4 K-tiles
  __shared__ __align__(16) _Float16 sW2[2 * 4 * 512];
  __shared__ float sB1[EMB], sB2[EMB];
  __shared__ __align__(16) _Float16 sHid[8][16 * EMB];

  const int tid = threadIdx.x;
  stage_w(w1, nullptr, 128, 4, sW1, tid, 256);
  stage_w(w2, nullptr, 64, 2, sW2, tid, 256);
  if (tid < EMB) { sB1[tid] = b1[tid]; sB2[tid] = b2[tid]; }
  __syncthreads();

  const int wv = tid >> 5, lane = tid & 31;
  const int lo = lane & 15, hi = lane >> 4;
  const int base = hi * 8;
  const int numTiles = NN / 16;

  for (int tile = blockIdx.x * 8 + wv; tile < numTiles; tile += gridDim.x * 8) {
    const _Float16* hrow = h16 + ((size_t)tile * 16 + lo) * EMB;
    const float*    arow = aggr + ((size_t)tile * 16 + lo) * EMB;

    v8f acc[4];
#pragma unroll
    for (int nt = 0; nt < 4; ++nt) {
      float bv = sB1[nt * 16 + lo];
#pragma unroll
      for (int r = 0; r < 8; ++r) acc[nt][r] = bv;
    }
#pragma unroll
    for (int kt = 0; kt < 2; ++kt) {                 // h part (K 0..63) from f16 shadow
      v16h a = load_a_f16row(hrow, kt * 32 + base);
#pragma unroll
      for (int nt = 0; nt < 4; ++nt)
        acc[nt] = wmma_f16(a, load_b_frag(&sW1[(size_t)(kt * 4 + nt) * 512 + lane * 16]), acc[nt]);
    }
#pragma unroll
    for (int kt = 0; kt < 2; ++kt) {                 // aggr part (K 64..127), f32 + cvt
      v16h a = load_a_f32row(arow, kt * 32 + base);
#pragma unroll
      for (int nt = 0; nt < 4; ++nt)
        acc[nt] = wmma_f16(a, load_b_frag(&sW1[(size_t)((kt + 2) * 4 + nt) * 512 + lane * 16]), acc[nt]);
    }

#pragma unroll
    for (int nt = 0; nt < 4; ++nt)
#pragma unroll
      for (int rp = 0; rp < 4; ++rp) {
        v2h p = relu_pair_f16(acc[nt][2 * rp], acc[nt][2 * rp + 1]);
        sHid[wv][(2 * rp     + 8 * hi) * EMB + nt * 16 + lo] = p[0];
        sHid[wv][(2 * rp + 1 + 8 * hi) * EMB + nt * 16 + lo] = p[1];
      }

    v8f acc2[4];
#pragma unroll
    for (int nt = 0; nt < 4; ++nt) {
      float bv = sB2[nt * 16 + lo];
#pragma unroll
      for (int r = 0; r < 8; ++r) acc2[nt][r] = bv;
    }
#pragma unroll
    for (int kt = 0; kt < 2; ++kt) {
      v16h a = load_a_f16row(&sHid[wv][lo * EMB], kt * 32 + base);
#pragma unroll
      for (int nt = 0; nt < 4; ++nt)
        acc2[nt] = wmma_f16(a, load_b_frag(&sW2[(size_t)(kt * 4 + nt) * 512 + lane * 16]), acc2[nt]);
    }

    // ReLU + residual; refresh f16 shadow (rows owned exclusively by this tile)
#pragma unroll
    for (int r = 0; r < 8; ++r) {
      size_t rowoff = ((size_t)tile * 16 + (r + 8 * hi)) * EMB;
      float*     orow = h   + rowoff;
      _Float16*  srow = h16 + rowoff;
#pragma unroll
      for (int nt = 0; nt < 4; ++nt) {
        int c = nt * 16 + lo;
        float v = orow[c] + relu(acc2[nt][r]);
        orow[c] = v;
        srow[c] = (_Float16)v;
      }
    }
  }
}

// ---------------------------------------------------------------------------
// Small kernels
// ---------------------------------------------------------------------------
__global__ __launch_bounds__(256) void lin_in_kernel(
    const float* __restrict__ x, const float* __restrict__ w,
    const float* __restrict__ b, float* __restrict__ h,
    _Float16* __restrict__ h16) {
  int n = blockIdx.x * blockDim.x + threadIdx.x;
  if (n >= NN) return;
  float xv[11];
#pragma unroll
  for (int k = 0; k < 11; ++k) xv[k] = x[(size_t)n * 11 + k];
#pragma unroll 4
  for (int c = 0; c < EMB; ++c) {
    float s = b[c];
#pragma unroll
    for (int k = 0; k < 11; ++k) s += xv[k] * w[k * EMB + c];
    h[(size_t)n * EMB + c] = s;
    h16[(size_t)n * EMB + c] = (_Float16)s;
  }
}

__global__ __launch_bounds__(256) void pred_kernel(
    const float* __restrict__ h, const float* __restrict__ w,
    const float* __restrict__ b, float* __restrict__ out) {
  int n = blockIdx.x * blockDim.x + threadIdx.x;
  if (n >= NN) return;
  float s = b[0];
#pragma unroll 8
  for (int k = 0; k < EMB; ++k) s += h[(size_t)n * EMB + k] * w[k];
  out[n] = s;
}

__global__ __launch_bounds__(256) void zero_kernel(float4* __restrict__ p, int n4) {
  int i = blockIdx.x * blockDim.x + threadIdx.x;
  if (i < n4) p[i] = make_float4(0.f, 0.f, 0.f, 0.f);
}

// ---------------------------------------------------------------------------
extern "C" void kernel_launch(void* const* d_in, const int* in_sizes, int n_in,
                              void* d_out, int out_size, void* d_ws, size_t ws_size,
                              hipStream_t stream) {
  const float*     x     = (const float*)d_in[0];
  const long long* ei    = (const long long*)d_in[1];   // int64 edge_index [2, E]
  const float*     ea    = (const float*)d_in[2];
  const float*     lin_w = (const float*)d_in[3];
  const float*     lin_b = (const float*)d_in[4];
  const float*     mw1   = (const float*)d_in[5];
  const float*     mb1   = (const float*)d_in[6];
  const float*     mw2   = (const float*)d_in[7];
  const float*     mb2   = (const float*)d_in[8];
  const float*     uw1   = (const float*)d_in[9];
  const float*     ub1   = (const float*)d_in[10];
  const float*     uw2   = (const float*)d_in[11];
  const float*     ub2   = (const float*)d_in[12];
  const float*     pw    = (const float*)d_in[13];
  const float*     pb    = (const float*)d_in[14];

  float*    h    = (float*)d_ws;                              // [N, 64] f32
  float*    aggr = h + (size_t)NN * EMB;                      // [N, 64] f32
  _Float16* h16  = (_Float16*)(aggr + (size_t)NN * EMB);      // [N, 64] f16 shadow

  lin_in_kernel<<<(NN + 255) / 256, 256, 0, stream>>>(x, lin_w, lin_b, h, h16);

  for (int l = 0; l < LAYERS; ++l) {
    zero_kernel<<<(NN * EMB / 4 + 255) / 256, 256, 0, stream>>>(
        (float4*)aggr, NN * EMB / 4);
    msg_kernel<<<1024, 256, 0, stream>>>(
        h16, ei, ea,
        mw1 + (size_t)l * 132 * EMB, mb1 + (size_t)l * EMB,
        mw2 + (size_t)l * EMB * EMB, mb2 + (size_t)l * EMB, aggr);
    upd_kernel<<<512, 256, 0, stream>>>(
        h, h16, aggr,
        uw1 + (size_t)l * 128 * EMB, ub1 + (size_t)l * EMB,
        uw2 + (size_t)l * EMB * EMB, ub2 + (size_t)l * EMB);
  }

  pred_kernel<<<(NN + 255) / 256, 256, 0, stream>>>(h, pw, pb, (float*)d_out);
}
